// Encoder_15461882265790
// MI455X (gfx1250) — compile-verified, matrix operand order checked by reading, not run on hardware
//
#include <hip/hip_runtime.h>

// ---------------------------------------------------------------------------
// Types for CDNA5 WMMA (wave32)
// ---------------------------------------------------------------------------
typedef __bf16 bf16x8  __attribute__((ext_vector_type(8)));    // 16 bytes
typedef __bf16 bf16x16 __attribute__((ext_vector_type(16)));   // 32 bytes
typedef float  f32x8   __attribute__((ext_vector_type(8)));

__device__ __forceinline__ unsigned short bf16_rne(float f) {
  unsigned int u = __builtin_bit_cast(unsigned int, f);
  u += 0x7FFFu + ((u >> 16) & 1u);           // round-to-nearest-even
  return (unsigned short)(u >> 16);
}

// ---------------------------------------------------------------------------
// C[M,N] = A[M,K] @ B[K,N], with A in bf16 [M,K] and B pre-transposed to
// bf16 Bt[N,K]. One wave32 computes one 16x16 tile via
// V_WMMA_F32_16X16X32_BF16. Per K-step: 4x global_load_b128 + 1x wmma.
//
// ISA 7.12.2 16-bit fragment packing: lane (half = lane>>4, mr = lane&15)
// holds two contiguous K-runs: [k0 + 8*half, +8) and [k0 + 16 + 8*half, +8).
// Requires M,N % 16 == 0 and K % 32 == 0 (holds: 100000 / 512,128 / 256,512).
// ---------------------------------------------------------------------------
__global__ void gemm_wmma_bf16(const __bf16* __restrict__ A,
                               const __bf16* __restrict__ Bt,
                               float* __restrict__ C,
                               int M, int N, int K) {
  const int lane   = threadIdx.x & 31;
  const int wave   = (int)((blockIdx.x * blockDim.x + threadIdx.x) >> 5);
  const int tilesN = N >> 4;
  const int tm     = wave / tilesN;
  const int tn     = wave - tm * tilesN;
  if (tm * 16 >= M) return;                  // wave-uniform: EXEC stays all-1s

  const int half = lane >> 4;
  const int mr   = lane & 15;
  const __bf16* __restrict__ arow = A  + (size_t)(tm * 16 + mr) * K + 8 * half;
  const __bf16* __restrict__ brow = Bt + (size_t)(tn * 16 + mr) * K + 8 * half;

  f32x8 acc = {};
#pragma unroll 4
  for (int k0 = 0; k0 < K; k0 += 32) {
    const bf16x8 alo = *(const bf16x8*)(arow + k0);
    const bf16x8 ahi = *(const bf16x8*)(arow + k0 + 16);
    const bf16x8 blo = *(const bf16x8*)(brow + k0);
    const bf16x8 bhi = *(const bf16x8*)(brow + k0 + 16);
    const bf16x16 av = __builtin_shufflevector(alo, ahi,
        0,1,2,3,4,5,6,7,8,9,10,11,12,13,14,15);
    const bf16x16 bv = __builtin_shufflevector(blo, bhi,
        0,1,2,3,4,5,6,7,8,9,10,11,12,13,14,15);
    // 8 args: (neg_a, A, neg_b, B, c_mod, C, reuse_a, reuse_b)
    acc = __builtin_amdgcn_wmma_f32_16x16x32_bf16(
        false, av, false, bv, (short)0, acc, false, false);
  }

  const int cn = tn * 16 + mr;
#pragma unroll
  for (int r = 0; r < 8; ++r)                // D: VGPR r -> row (r + 8*half)
    C[(size_t)(tm * 16 + r + 8 * half) * N + cn] = acc[r];
}

// ---------------------------------------------------------------------------
// Elementwise f32 -> bf16 (4-wide: float4 load, 8B store)
// ---------------------------------------------------------------------------
__global__ void cvt_f32_bf16_x4(const float* __restrict__ in,
                                unsigned short* __restrict__ out,
                                long long n4) {
  long long i = (long long)blockIdx.x * blockDim.x + threadIdx.x;
  if (i >= n4) return;
  const float4 v = ((const float4*)in)[i];
  ushort4 o;
  o.x = bf16_rne(v.x); o.y = bf16_rne(v.y);
  o.z = bf16_rne(v.z); o.w = bf16_rne(v.w);
  ((ushort4*)out)[i] = o;
}

// ---------------------------------------------------------------------------
// W[K,N] f32 -> Wt[N,K] bf16 (transpose; coalesced reads, tiny matrix)
// ---------------------------------------------------------------------------
__global__ void cvt_w_transpose(const float* __restrict__ W,
                                unsigned short* __restrict__ Wt,
                                int K, int N) {
  long long i = (long long)blockIdx.x * blockDim.x + threadIdx.x;
  if (i >= (long long)K * N) return;
  const int k = (int)(i / N);
  const int n = (int)(i - (long long)k * N);
  Wt[(size_t)n * K + k] = bf16_rne(W[i]);
}

// ---------------------------------------------------------------------------
// Degree / normalization
// ---------------------------------------------------------------------------
__global__ void deg_init(float* __restrict__ deg, int n) {
  int i = blockIdx.x * blockDim.x + threadIdx.x;
  if (i < n) deg[i] = 1.0f;                  // self-loop contribution
}

__global__ void deg_count(const long long* __restrict__ dst,
                          float* __restrict__ deg, int e) {
  int i = blockIdx.x * blockDim.x + threadIdx.x;
  if (i < e) atomicAdd(&deg[dst[i]], 1.0f);
}

__global__ void deg_rsqrt(float* __restrict__ deg, int n) {
  int i = blockIdx.x * blockDim.x + threadIdx.x;
  if (i < n) deg[i] = rsqrtf(deg[i]);        // deg >= 1, becomes dis
}

// ---------------------------------------------------------------------------
// out[n][c] = bias[c] + dis[n]^2 * h[n][c]  (bias + self-loop, init accum)
// ---------------------------------------------------------------------------
__global__ void init_out(const float* __restrict__ h,
                         const float* __restrict__ dis,
                         const float* __restrict__ bias,
                         float* __restrict__ out,
                         long long total, int cshift) {
  long long i = (long long)blockIdx.x * blockDim.x + threadIdx.x;
  if (i >= total) return;
  const int   node = (int)(i >> cshift);
  const int   c    = (int)(i & ((1 << cshift) - 1));
  const float d    = dis[node];
  out[i] = bias[c] + d * d * h[i];
}

// ---------------------------------------------------------------------------
// Edge scatter: out[dst] += h[src] * dis[src]*dis[dst]
// One lane per float4 chunk; lanes of a wave share an edge row.
// ---------------------------------------------------------------------------
__global__ void scatter_edges(const float* __restrict__ h,
                              const float* __restrict__ dis,
                              const long long* __restrict__ src,
                              const long long* __restrict__ dst,
                              float* __restrict__ out,
                              long long total, int c4shift, int C) {
  long long gid = (long long)blockIdx.x * blockDim.x + threadIdx.x;
  if (gid >= total) return;
  const int e     = (int)(gid >> c4shift);
  const int chunk = (int)(gid & ((1 << c4shift) - 1));
  const long long s = src[e];
  const long long d = dst[e];
  const float nrm = dis[s] * dis[d];
  const float4 v = ((const float4*)(h + (size_t)s * C))[chunk];
  float* op = out + (size_t)d * C + (size_t)chunk * 4;
  atomicAdd(op + 0, v.x * nrm);
  atomicAdd(op + 1, v.y * nrm);
  atomicAdd(op + 2, v.z * nrm);
  atomicAdd(op + 3, v.w * nrm);
}

__global__ void relu_ip(float* __restrict__ p, long long n) {
  long long i = (long long)blockIdx.x * blockDim.x + threadIdx.x;
  if (i < n) p[i] = fmaxf(p[i], 0.0f);
}

// ---------------------------------------------------------------------------
// Host-side launch
// ---------------------------------------------------------------------------
static inline unsigned cdiv_ll(long long a, long long b) {
  return (unsigned)((a + b - 1) / b);
}

extern "C" void kernel_launch(void* const* d_in, const int* in_sizes, int n_in,
                              void* d_out, int out_size, void* d_ws, size_t ws_size,
                              hipStream_t stream) {
  (void)n_in; (void)out_size; (void)ws_size;

  const float*     x   = (const float*)d_in[0];
  const long long* ei  = (const long long*)d_in[1];   // int64 [2, E]
  const float*     W1  = (const float*)d_in[2];
  const float*     b1  = (const float*)d_in[3];
  const float*     W2  = (const float*)d_in[4];
  const float*     b2  = (const float*)d_in[5];
  float*           out = (float*)d_out;

  const int IN  = 256;
  const int N   = in_sizes[0] / IN;        // 100000
  const int E   = in_sizes[1] / 2;         // 1600000
  const int HID = in_sizes[2] / IN;        // 512
  const int OUT = in_sizes[4] / HID;       // 128
  const int HSH = 9;                       // log2(HID)
  const int OSH = 7;                       // log2(OUT)

  const long long* src = ei;
  const long long* dst = ei + E;

  // ---- workspace layout (bytes), 256B-aligned regions ----
  char* p = (char*)d_ws;
  auto align256 = [](size_t b) { return (b + 255) & ~(size_t)255; };
  float*          dis = (float*)p;                 p += align256((size_t)N * 4);
  unsigned short* W1t = (unsigned short*)p;        p += align256((size_t)IN * HID * 2);
  unsigned short* W2t = (unsigned short*)p;        p += align256((size_t)HID * OUT * 2);
  float*          h1  = (float*)p;                 p += align256((size_t)N * HID * 4);
  float*          o1  = (float*)p;               /*p += align256((size_t)N * HID * 4);*/
  // aliases (stream-ordered lifetimes, see comments):
  unsigned short* xb  = (unsigned short*)o1;       // bf16(x): dead before o1 written
  unsigned short* o1b = (unsigned short*)h1;       // bf16(o1): h1 dead after scatter1
  float*          h2  = (float*)((char*)h1 + align256((size_t)N * HID * 2)); // after o1b

  const int TB = 256;

  // --- degree normalization: dis = rsqrt(1 + in-degree) ---
  deg_init <<<cdiv_ll(N, TB), TB, 0, stream>>>(dis, N);
  deg_count<<<cdiv_ll(E, TB), TB, 0, stream>>>(dst, dis, E);
  deg_rsqrt<<<cdiv_ll(N, TB), TB, 0, stream>>>(dis, N);

  // --- one-time weight conversion + transpose (bf16) ---
  cvt_w_transpose<<<cdiv_ll((long long)IN * HID, TB), TB, 0, stream>>>(W1, W1t, IN, HID);
  cvt_w_transpose<<<cdiv_ll((long long)HID * OUT, TB), TB, 0, stream>>>(W2, W2t, HID, OUT);

  // --- layer 1: h1 = x @ W1 ; o1 = relu(aggregate + b1) ---
  cvt_f32_bf16_x4<<<cdiv_ll((long long)N * IN / 4, TB), TB, 0, stream>>>(
      x, xb, (long long)N * IN / 4);
  {
    const long long waves = (long long)((N + 15) / 16) * (HID / 16);
    gemm_wmma_bf16<<<cdiv_ll(waves * 32, TB), TB, 0, stream>>>(
        (const __bf16*)xb, (const __bf16*)W1t, h1, N, HID, IN);
  }
  {
    const long long tot = (long long)N * HID;
    init_out<<<cdiv_ll(tot, TB), TB, 0, stream>>>(h1, dis, b1, o1, tot, HSH);
    const long long etot = (long long)E * (HID / 4);
    scatter_edges<<<cdiv_ll(etot, TB), TB, 0, stream>>>(h1, dis, src, dst, o1,
                                                        etot, HSH - 2, HID);
    relu_ip<<<cdiv_ll(tot, TB), TB, 0, stream>>>(o1, tot);
  }

  // --- layer 2: h2 = o1 @ W2 ; out = relu(aggregate + b2) ---
  cvt_f32_bf16_x4<<<cdiv_ll((long long)N * HID / 4, TB), TB, 0, stream>>>(
      o1, o1b, (long long)N * HID / 4);
  {
    const long long waves = (long long)((N + 15) / 16) * (OUT / 16);
    gemm_wmma_bf16<<<cdiv_ll(waves * 32, TB), TB, 0, stream>>>(
        (const __bf16*)o1b, (const __bf16*)W2t, h2, N, OUT, HID);
  }
  {
    const long long tot = (long long)N * OUT;
    init_out<<<cdiv_ll(tot, TB), TB, 0, stream>>>(h2, dis, b2, out, tot, OSH);
    const long long etot = (long long)E * (OUT / 4);
    scatter_edges<<<cdiv_ll(etot, TB), TB, 0, stream>>>(h2, dis, src, dst, out,
                                                        etot, OSH - 2, OUT);
    relu_ip<<<cdiv_ll(tot, TB), TB, 0, stream>>>(out, tot);
  }
}
